// Direct_Note_74165495267680
// MI455X (gfx1250) — matrix operand order repaired and checked by reading, not compiled, not run on hardware
//
#include <hip/hip_runtime.h>
#include <hip/hip_bf16.h>

// ---------------------------------------------------------------------------
// Types for CDNA5 WMMA (wave32, 16x16x32 f16 -> f32)
// ---------------------------------------------------------------------------
typedef _Float16 v16h __attribute__((ext_vector_type(16)));
typedef _Float16 v8h  __attribute__((ext_vector_type(8)));
typedef float    v8f  __attribute__((ext_vector_type(8)));

__device__ __forceinline__ float sigf(float x) { return 1.0f / (1.0f + expf(-x)); }

// A 16x32 f16 fragment: lanes 0-15 K in {0..7,16..23}; lanes 16-31 K in {8..15,24..31}
__device__ __forceinline__ v16h load_a_frag(const _Float16* __restrict__ rowBase, int k0, int half) {
  const _Float16* p = rowBase + k0 + half * 8;
  v8h lo = *reinterpret_cast<const v8h*>(p);
  v8h hi = *reinterpret_cast<const v8h*>(p + 16);
  return __builtin_shufflevector(lo, hi, 0, 1, 2, 3, 4, 5, 6, 7, 8, 9, 10, 11, 12, 13, 14, 15);
}
// B 32x16 f16 fragment: lanes 0-15 K in {0..7,16..23}; lanes 16-31 K in {8..15,24..31}
__device__ __forceinline__ v16h load_b_frag(const _Float16* __restrict__ rowBase, int k0, int half) {
  const _Float16* p = rowBase + k0 + half * 8;
  v8h lo = *reinterpret_cast<const v8h*>(p);
  v8h hi = *reinterpret_cast<const v8h*>(p + 16);
  return __builtin_shufflevector(lo, hi, 0, 1, 2, 3, 4, 5, 6, 7, 8, 9, 10, 11, 12, 13, 14, 15);
}

// ---------------------------------------------------------------------------
// f32 -> f16 convert with column zero-padding (for WMMA operands)
// ---------------------------------------------------------------------------
__global__ void cvt_pad_f16_kernel(const float* __restrict__ src, _Float16* __restrict__ dst,
                                   long long rows, int cols, int colsPad) {
  long long i = (long long)blockIdx.x * blockDim.x + threadIdx.x;
  long long total = rows * (long long)colsPad;
  if (i >= total) return;
  long long r = i / colsPad;
  int c = (int)(i - r * colsPad);
  dst[i] = (c < cols) ? (_Float16)src[r * cols + c] : (_Float16)0.0f;
}

// ---------------------------------------------------------------------------
// WMMA GEMM: C[M,N] = act(A[M,K] * W[N,K]^T + bias + bias2)
// A row m read at A[(aRow0 + m*aRowStep)*lda + k]  (f16, lda mult of 8, K mult of 32)
// W row-major [N, ldb] f16.  M, N multiples of 32.
// One wave computes a 32x32 output block (2x2 register-blocked 16x16 WMMA tiles):
// 4 v_wmma per K-step sharing 2 A-frags + 2 B-frags -> 2x the FLOP/byte of 1 tile/wave.
// Output mapping: idx = (cRow0 + gm*cRowStep)*ldc + cCol0 + gn  (f32 Cf and/or f16 Ch)
// ---------------------------------------------------------------------------
__global__ __launch_bounds__(256) void wmma_gemm_kernel(
    const _Float16* __restrict__ A, long long aRow0, int aRowStep, int lda,
    const _Float16* __restrict__ Bw, int ldb,
    const float* __restrict__ bias, const float* __restrict__ bias2,
    float* __restrict__ Cf, _Float16* __restrict__ Ch,
    long long cRow0, int cRowStep, int ldc, int cCol0,
    int M, int N, int K, int act) {
  const int lane = threadIdx.x & 31;
  const int half = lane >> 4;      // selects K sub-range per WMMA layout
  const int nl   = lane & 15;      // row (A) / col (B,C) index within 16x16 tile
  const int wavesPerBlk = blockDim.x >> 5;
  const int wave = blockIdx.x * wavesPerBlk + (threadIdx.x >> 5);
  const int tilesN = N >> 5;
  const int numTiles = (M >> 5) * tilesN;
  const int nWaves = gridDim.x * wavesPerBlk;
  for (int tile = wave; tile < numTiles; tile += nWaves) {
    const int tm = tile / tilesN;
    const int tn = tile - tm * tilesN;
    const _Float16* A0 = A + (aRow0 + (long long)(tm * 32 + nl) * aRowStep) * (long long)lda;
    const _Float16* A1 = A + (aRow0 + (long long)(tm * 32 + 16 + nl) * aRowStep) * (long long)lda;
    const _Float16* B0 = Bw + (long long)(tn * 32 + nl) * ldb;
    const _Float16* B1 = Bw + (long long)(tn * 32 + 16 + nl) * ldb;
    v8f acc00 = {}, acc01 = {}, acc10 = {}, acc11 = {};
    for (int k0 = 0; k0 < K; k0 += 32) {
      v16h a0 = load_a_frag(A0, k0, half);
      v16h a1 = load_a_frag(A1, k0, half);
      v16h b0 = load_b_frag(B0, k0, half);
      v16h b1 = load_b_frag(B1, k0, half);
      acc00 = __builtin_amdgcn_wmma_f32_16x16x32_f16(false, a0, false, b0, (short)0, acc00, false, false);
      acc01 = __builtin_amdgcn_wmma_f32_16x16x32_f16(false, a0, false, b1, (short)0, acc01, false, false);
      acc10 = __builtin_amdgcn_wmma_f32_16x16x32_f16(false, a1, false, b0, (short)0, acc10, false, false);
      acc11 = __builtin_amdgcn_wmma_f32_16x16x32_f16(false, a1, false, b1, (short)0, acc11, false, false);
    }
    // store 2x2 sub-tiles; C layout: VGPR r holds (M = r + half*8, N = nl) of each 16x16 tile
#pragma unroll
    for (int sub = 0; sub < 4; ++sub) {
      const int sm = sub >> 1, sn = sub & 1;
      v8f acc = (sub == 0) ? acc00 : (sub == 1) ? acc01 : (sub == 2) ? acc10 : acc11;
      const int gn = tn * 32 + sn * 16 + nl;
      float badd = 0.0f;
      if (bias)  badd += bias[gn];
      if (bias2) badd += bias2[gn];
#pragma unroll
      for (int r = 0; r < 8; ++r) {
        const int gm = tm * 32 + sm * 16 + r + half * 8;
        float v = acc[r] + badd;
        if (act == 1) v = v > 0.0f ? v : 0.0f;
        else if (act == 2) v = tanhf(v);
        const long long row = cRow0 + (long long)gm * cRowStep;
        const long long o = row * ldc + cCol0 + gn;
        if (Cf) Cf[o] = v;
        if (Ch) Ch[o] = (_Float16)v;
      }
    }
  }
}

// ---------------------------------------------------------------------------
// LSTM scan: gates precomputed as gih[(seq*2+dir)][T][4H] = x_t @ w_ih^T + b_ih + b_hh.
// One block per (seq, dir). h,c live in LDS; w_hh matvec from global (L2-resident).
// Output row = outRow0 + seq*outSeqStride + t*outRowStep, col = colBase + dir*H + j.
// ---------------------------------------------------------------------------
__global__ __launch_bounds__(512) void lstm_scan_kernel(
    const float* __restrict__ gih,
    const float* __restrict__ whh_f, const float* __restrict__ whh_b,
    float* __restrict__ outF, _Float16* __restrict__ outH,
    int T, int H, int ldOut, int colBase,
    long long outRow0, long long outSeqStride, long long outRowStep) {
  const int blk = blockIdx.x;
  const int seq = blk >> 1, dir = blk & 1;
  const int G4 = 4 * H;
  const float* gblock = gih + (long long)blk * T * G4;
  const float* whh = dir ? whh_b : whh_f;
  __shared__ float sh[512];
  __shared__ float sc[512];
  __shared__ float sg[2048];
  const int tid = threadIdx.x;
  for (int j = tid; j < H; j += blockDim.x) { sh[j] = 0.0f; sc[j] = 0.0f; }
  __syncthreads();
  const int col0 = colBase + dir * H;
  for (int step = 0; step < T; ++step) {
    const int tt = dir ? (T - 1 - step) : step;
    const float* grow = gblock + (long long)tt * G4;
    for (int j = tid; j < G4; j += blockDim.x) {
      const float* wr = whh + (long long)j * H;
      float acc = grow[j];
      for (int k = 0; k < H; ++k) acc += wr[k] * sh[k];
      sg[j] = acc;
    }
    __syncthreads();
    for (int j = tid; j < H; j += blockDim.x) {
      float iv = sigf(sg[j]);
      float fv = sigf(sg[H + j]);
      float gv = tanhf(sg[2 * H + j]);
      float ov = sigf(sg[3 * H + j]);
      float c = fv * sc[j] + iv * gv;
      float h = ov * tanhf(c);
      sc[j] = c; sh[j] = h;
      long long row = outRow0 + (long long)seq * outSeqStride + (long long)tt * outRowStep;
      long long o = row * ldOut + col0 + j;
      if (outF) outF[o] = h;
      if (outH) outH[o] = (_Float16)h;
    }
    __syncthreads();
  }
}

// ---------------------------------------------------------------------------
// Context attention over groups of G rows: a = tanh(xW^T+b) precomputed.
// sim[t,h] = <a[t,h*hs:], ctx[h]>, softmax over t, out = sum_t w * x.
// ---------------------------------------------------------------------------
__global__ void group_attention_kernel(const float* __restrict__ x, const float* __restrict__ a,
                                       const float* __restrict__ ctx,
                                       float* __restrict__ outF, _Float16* __restrict__ outH,
                                       int S, int nHead, int G) {
  const int g = blockIdx.x;
  const int hs = S / nHead;
  __shared__ float sim[64];
  __shared__ float wgt[64];
  const int tid = threadIdx.x;
  if (tid < G * nHead) {
    int t = tid / nHead, h_ = tid % nHead;
    const float* ap = a + (long long)(g * G + t) * S + h_ * hs;
    const float* cp = ctx + h_ * hs;
    float s_ = 0.0f;
    for (int d = 0; d < hs; ++d) s_ += ap[d] * cp[d];
    sim[t * nHead + h_] = s_;
  }
  __syncthreads();
  if (tid < nHead) {
    float mx = -1e30f;
    for (int t = 0; t < G; ++t) mx = fmaxf(mx, sim[t * nHead + tid]);
    float e[8]; float sm = 0.0f;
    for (int t = 0; t < G; ++t) { e[t] = expf(sim[t * nHead + tid] - mx); sm += e[t]; }
    for (int t = 0; t < G; ++t) wgt[t * nHead + tid] = e[t] / sm;
  }
  __syncthreads();
  for (int c = tid; c < S; c += blockDim.x) {
    int h_ = c / hs;
    float acc = 0.0f;
    for (int t = 0; t < G; ++t) acc += wgt[t * nHead + h_] * x[(long long)(g * G + t) * S + c];
    long long o = (long long)g * S + c;
    outF[o] = acc;
    if (outH) outH[o] = (_Float16)acc;
  }
}

// ---------------------------------------------------------------------------
// Decoder static-input builders (f16, dynamic columns zeroed)
// tempo cat (1555->1568): [beatH 1024 | measH 512 | prev_tempo | qpm | tp(2) | tvec(5) | rnode(10) | pad]
// note cat (2318->2336): [note 768 | beatH 1024 | measH 512 | p_out(11) | qpm | tp(2) | pad]
// ---------------------------------------------------------------------------
__global__ void build_tempo_static_kernel(const float* __restrict__ x, const float* __restrict__ beatH,
                                          const float* __restrict__ measH, _Float16* __restrict__ out) {
  const int W = 1568;
  long long i = (long long)blockIdx.x * blockDim.x + threadIdx.x;
  if (i >= 512LL * W) return;
  int b = (int)(i / W), c = (int)(i % W);
  float v = 0.0f;
  if (c < 1024) v = beatH[b * 1024 + c];
  else if (c < 1536) v = measH[(b >> 2) * 512 + (c - 1024)];
  else if (c == 1537) v = x[4];                       // qpm_primo
  else if (c == 1538 || c == 1539) v = x[5 + (c - 1538)];  // tempo_primo
  else if (c >= 1540 && c < 1545) v = x[(long long)(b * 4) * 78 + 26 + (c - 1540)];  // tvec
  out[i] = (_Float16)v;
}

__global__ void build_note_static_kernel(const float* __restrict__ x, const float* __restrict__ noteOut,
                                         const float* __restrict__ beatH, const float* __restrict__ measH,
                                         _Float16* __restrict__ out) {
  const int W = 2336;
  long long i = (long long)blockIdx.x * blockDim.x + threadIdx.x;
  if (i >= 2048LL * W) return;
  int n = (int)(i / W), c = (int)(i % W);
  float v = 0.0f;
  if (c < 768) v = noteOut[(long long)n * 768 + c];
  else if (c < 1792) v = beatH[(n >> 2) * 1024 + (c - 768)];
  else if (c < 2304) v = measH[(n >> 4) * 512 + (c - 1792)];
  else if (c == 2315) v = x[4];
  else if (c == 2316 || c == 2317) v = x[5 + (c - 2316)];
  out[i] = (_Float16)v;
}

// ---------------------------------------------------------------------------
// Autoregressive decoder: 512 beats x (tempo LSTM cell + 4 note steps of 2-layer LSTM).
// Static input projections were hoisted into WMMA GEMMs (gstatT/gstatN); the scan only
// adds the 11 dynamic columns + recurrent matvecs. Single persistent workgroup.
// ---------------------------------------------------------------------------
__global__ __launch_bounds__(1024) void decoder_scan_kernel(
    const float* __restrict__ gstatT,  // [512][2048]
    const float* __restrict__ wihT,    // [2048][1555]
    const float* __restrict__ whhT,    // [2048][512]
    const float* __restrict__ tfcW,    // [512]
    const float* __restrict__ tfcB,    // [1]
    const float* __restrict__ gstatN,  // [2048][1024]
    const float* __restrict__ wih0,    // [1024][2318]
    const float* __restrict__ whh0,    // [1024][256]
    const float* __restrict__ wih1,    // [1024][256]
    const float* __restrict__ whh1,    // [1024][256]
    const float* __restrict__ bih1, const float* __restrict__ bhh1,
    const float* __restrict__ ffcW,    // [10][256]
    const float* __restrict__ ffcB,    // [10]
    const float* __restrict__ taW,     // [10][10]
    const float* __restrict__ taB,     // [10]
    const float* __restrict__ taCtx,   // [10]
    float* __restrict__ outTotal) {    // [2048][11]
  __shared__ float s_th[512];
  __shared__ float s_tc[512];
  __shared__ float s_h1[256];
  __shared__ float s_c1[256];
  __shared__ float s_h2[256];
  __shared__ float s_c2[256];
  __shared__ float s_g[2048];
  __shared__ float s_p10[40];
  __shared__ float s_rn[10];
  __shared__ float s_pout[11];
  __shared__ float s_sim[4];
  __shared__ float s_tmp;
  const int tid = threadIdx.x;
  if (tid < 512) { s_th[tid] = 0.0f; s_tc[tid] = 0.0f; }
  if (tid < 256) { s_h1[tid] = 0.0f; s_c1[tid] = 0.0f; s_h2[tid] = 0.0f; s_c2[tid] = 0.0f; }
  if (tid < 40) s_p10[tid] = 0.0f;
  if (tid < 11) s_pout[tid] = 0.0f;
  if (tid == 0) s_tmp = 0.0f;
  __syncthreads();

  for (int b = 0; b < 512; ++b) {
    // ---- rnode = context_attention(prev10) (1 head, hs=10); zero for b==0 ----
    if (b == 0) {
      if (tid < 10) s_rn[tid] = 0.0f;
    } else {
      if (tid < 4) {
        float av[10]; float s = 0.0f;
        for (int d = 0; d < 10; ++d) {
          float z = taB[d];
          for (int e = 0; e < 10; ++e) z += taW[d * 10 + e] * s_p10[tid * 10 + e];
          av[d] = tanhf(z);
        }
        for (int d = 0; d < 10; ++d) s += av[d] * taCtx[d];
        s_sim[tid] = s;
      }
      __syncthreads();
      if (tid == 0) {
        float mx = fmaxf(fmaxf(s_sim[0], s_sim[1]), fmaxf(s_sim[2], s_sim[3]));
        float e0 = expf(s_sim[0] - mx), e1 = expf(s_sim[1] - mx);
        float e2 = expf(s_sim[2] - mx), e3 = expf(s_sim[3] - mx);
        float sm = e0 + e1 + e2 + e3;
        s_sim[0] = e0 / sm; s_sim[1] = e1 / sm; s_sim[2] = e2 / sm; s_sim[3] = e3 / sm;
      }
      __syncthreads();
      if (tid < 10) {
        float r = 0.0f;
        for (int t = 0; t < 4; ++t) r += s_sim[t] * s_p10[t * 10 + tid];
        s_rn[tid] = r;
      }
    }
    __syncthreads();
    const float pt = s_tmp;  // previous beat's tempo
    // ---- tempo LSTM gates: static + dynamic cols + w_hh @ th ----
    for (int j = tid; j < 2048; j += blockDim.x) {
      const float* wr = wihT + (long long)j * 1555;
      float g = gstatT[(long long)b * 2048 + j] + wr[1536] * pt;
      for (int d = 0; d < 10; ++d) g += wr[1545 + d] * s_rn[d];
      const float* hr = whhT + (long long)j * 512;
      float acc = 0.0f;
      for (int k = 0; k < 512; ++k) acc += hr[k] * s_th[k];
      s_g[j] = g + acc;
    }
    __syncthreads();
    if (tid < 512) {
      float iv = sigf(s_g[tid]), fv = sigf(s_g[512 + tid]);
      float gv = tanhf(s_g[1024 + tid]), ov = sigf(s_g[1536 + tid]);
      float c = fv * s_tc[tid] + iv * gv;
      float h = ov * tanhf(c);
      s_tc[tid] = c; s_th[tid] = h;
    }
    __syncthreads();
    // ---- prev_tempo = tfc(th) (tree reduce) ----
    if (tid < 512) s_g[tid] = tfcW[tid] * s_th[tid];
    __syncthreads();
    for (int off = 256; off > 0; off >>= 1) {
      if (tid < off) s_g[tid] += s_g[tid + off];
      __syncthreads();
    }
    if (tid == 0) s_tmp = s_g[0] + tfcB[0];
    __syncthreads();
    const float ptempo = s_tmp;
    // ---- 4 note steps ----
    for (int t = 0; t < 4; ++t) {
      const int n = b * 4 + t;
      for (int j = tid; j < 1024; j += blockDim.x) {
        const float* wr = wih0 + (long long)j * 2318;
        float g = gstatN[(long long)n * 1024 + j];
        for (int d = 0; d < 11; ++d) g += wr[2304 + d] * s_pout[d];
        const float* hr = whh0 + (long long)j * 256;
        float acc = 0.0f;
        for (int k = 0; k < 256; ++k) acc += hr[k] * s_h1[k];
        s_g[j] = g + acc;
      }
      __syncthreads();
      if (tid < 256) {
        float iv = sigf(s_g[tid]), fv = sigf(s_g[256 + tid]);
        float gv = tanhf(s_g[512 + tid]), ov = sigf(s_g[768 + tid]);
        float c = fv * s_c1[tid] + iv * gv;
        float h = ov * tanhf(c);
        s_c1[tid] = c; s_h1[tid] = h;
      }
      __syncthreads();
      for (int j = tid; j < 1024; j += blockDim.x) {
        float g = bih1[j] + bhh1[j];
        const float* ir = wih1 + (long long)j * 256;
        const float* hr = whh1 + (long long)j * 256;
        float acc = 0.0f;
        for (int k = 0; k < 256; ++k) acc += ir[k] * s_h1[k] + hr[k] * s_h2[k];
        s_g[j] = g + acc;
      }
      __syncthreads();
      if (tid < 256) {
        float iv = sigf(s_g[tid]), fv = sigf(s_g[256 + tid]);
        float gv = tanhf(s_g[512 + tid]), ov = sigf(s_g[768 + tid]);
        float c = fv * s_c2[tid] + iv * gv;
        float h = ov * tanhf(c);
        s_c2[tid] = c; s_h2[tid] = h;
      }
      __syncthreads();
      if (tid < 10) {
        float o = ffcB[tid];
        const float* fr = ffcW + tid * 256;
        for (int k = 0; k < 256; ++k) o += fr[k] * s_h2[k];
        outTotal[(long long)n * 11 + 1 + tid] = o;
        s_p10[t * 10 + tid] = o;
        s_pout[1 + tid] = o;
      }
      if (tid == 0) {
        outTotal[(long long)n * 11] = ptempo;
        s_pout[0] = ptempo;
      }
      __syncthreads();
    }
  }
}

// ---------------------------------------------------------------------------
// hidden_total = concat(note_out[n], beat_hidden[bn[n]], measure_hidden[mn[n]])
// ---------------------------------------------------------------------------
__global__ void hidden_total_kernel(const float* __restrict__ noteOut, const float* __restrict__ beatH,
                                    const float* __restrict__ measH, const int* __restrict__ bn,
                                    const int* __restrict__ mn, float* __restrict__ out) {
  long long i = (long long)blockIdx.x * blockDim.x + threadIdx.x;
  if (i >= 2048LL * 2304) return;
  int n = (int)(i / 2304), c = (int)(i % 2304);
  float v;
  if (c < 768) v = noteOut[(long long)n * 768 + c];
  else if (c < 1792) v = beatH[(long long)bn[n] * 1024 + (c - 768)];
  else v = measH[(long long)mn[n] * 512 + (c - 1792)];
  out[i] = v;
}

// ---------------------------------------------------------------------------
// Host orchestration
// ---------------------------------------------------------------------------
static void launch_gemm(hipStream_t s, const _Float16* A, long long aRow0, int aRowStep, int lda,
                        const _Float16* B, int ldb, const float* bias, const float* bias2,
                        float* Cf, _Float16* Ch, long long cRow0, int cRowStep, int ldc, int cCol0,
                        int M, int N, int K, int act) {
  int tiles = (M / 32) * (N / 32);
  int blocks = (tiles + 7) / 8;
  if (blocks > 2048) blocks = 2048;
  if (blocks < 1) blocks = 1;
  wmma_gemm_kernel<<<blocks, 256, 0, s>>>(A, aRow0, aRowStep, lda, B, ldb, bias, bias2,
                                          Cf, Ch, cRow0, cRowStep, ldc, cCol0, M, N, K, act);
}

static void launch_cvt(hipStream_t s, const float* src, _Float16* dst, long long rows, int cols, int colsPad) {
  long long total = rows * (long long)colsPad;
  int blocks = (int)((total + 255) / 256);
  cvt_pad_f16_kernel<<<blocks, 256, 0, s>>>(src, dst, rows, cols, colsPad);
}

extern "C" void kernel_launch(void* const* d_in, const int* in_sizes, int n_in,
                              void* d_out, int out_size, void* d_ws, size_t ws_size,
                              hipStream_t stream) {
  (void)in_sizes; (void)n_in; (void)out_size; (void)ws_size;
  auto F = [&](int i) { return (const float*)d_in[i]; };

  // ---- workspace bump allocator ----
  char* wp = (char*)d_ws;
  auto alloc = [&](size_t bytes) -> void* {
    void* r = (void*)wp;
    wp += (bytes + 255) & ~(size_t)255;
    return r;
  };
  auto a16 = [&](size_t elems) { return (_Float16*)alloc(elems * sizeof(_Float16)); };
  auto a32 = [&](size_t elems) { return (float*)alloc(elems * sizeof(float)); };

  // f16 buffers
  _Float16* Xp16     = a16(2048 * 96);
  _Float16* W0p      = a16(256 * 96);
  _Float16* W1p      = a16(256 * 256);
  _Float16* W2p      = a16(256 * 256);
  _Float16* Hfc1     = a16(2048 * 256);
  _Float16* Hfc2     = a16(2048 * 256);
  _Float16* H16      = a16(2048 * 256);
  _Float16* vwih[4]  = { a16(512 * 256), a16(512 * 256), a16(512 * 256), a16(512 * 256) }; // l0f,l0b,l1f,l1b
  _Float16* lwih0[2] = { a16(1024 * 256), a16(1024 * 256) };
  _Float16* lwih1[2] = { a16(1024 * 512), a16(1024 * 512) };
  _Float16* vbuf0    = a16(4 * 512 * 256);
  _Float16* l0out16  = a16(2048 * 512);
  _Float16* noteOut16= a16(2048 * 768);
  _Float16* battnW16 = a16(768 * 768);
  _Float16* bnodes16 = a16(512 * 768);
  _Float16* bwih0[2] = { a16(2048 * 768), a16(2048 * 768) };
  _Float16* bwih1[2] = { a16(2048 * 1024), a16(2048 * 1024) };
  _Float16* beatl016 = a16(512 * 1024);
  _Float16* beatH16  = a16(512 * 1024);
  _Float16* mattnW16 = a16(1024 * 1024);
  _Float16* mnodes16 = a16(128 * 1024);
  _Float16* mwih[2]  = { a16(1024 * 1024), a16(1024 * 1024) };
  _Float16* tStat16  = a16(512 * 1568);
  _Float16* wihT16   = a16(2048 * 1568);
  _Float16* nStat16  = a16(2048 * 2336);
  _Float16* wih0_16  = a16(1024 * 2336);
  // f32 buffers
  float* Gih      = a32(2 * 2048 * 1024);  // reused for every bi-LSTM layer
  float* noteOutF = a32(2048 * 768);
  float* attnA    = a32(2048 * 768);
  float* bnodesF  = a32(512 * 768);
  float* beatHF   = a32(512 * 1024);
  float* mnodesF  = a32(128 * 1024);
  float* measHF   = a32(128 * 512);
  float* gstatT   = a32(512 * 2048);
  float* gstatN   = a32(2048 * 1024);

  // ---- weight / input conversions to f16 (K padded to mult of 32) ----
  launch_cvt(stream, F(0),  Xp16,   2048, 78,  96);
  launch_cvt(stream, F(6),  W0p,    256,  78,  96);
  launch_cvt(stream, F(8),  W1p,    256,  256, 256);
  launch_cvt(stream, F(10), W2p,    256,  256, 256);
  const int vwidx[4] = { 12, 16, 20, 24 };
  for (int i = 0; i < 4; ++i) launch_cvt(stream, F(vwidx[i]), vwih[i], 512, 256, 256);
  launch_cvt(stream, F(28), lwih0[0], 1024, 256, 256);
  launch_cvt(stream, F(32), lwih0[1], 1024, 256, 256);
  launch_cvt(stream, F(36), lwih1[0], 1024, 512, 512);
  launch_cvt(stream, F(40), lwih1[1], 1024, 512, 512);
  launch_cvt(stream, F(44), battnW16, 768, 768, 768);
  launch_cvt(stream, F(47), bwih0[0], 2048, 768, 768);
  launch_cvt(stream, F(51), bwih0[1], 2048, 768, 768);
  launch_cvt(stream, F(55), bwih1[0], 2048, 1024, 1024);
  launch_cvt(stream, F(59), bwih1[1], 2048, 1024, 1024);
  launch_cvt(stream, F(63), mattnW16, 1024, 1024, 1024);
  launch_cvt(stream, F(66), mwih[0], 1024, 1024, 1024);
  launch_cvt(stream, F(70), mwih[1], 1024, 1024, 1024);
  launch_cvt(stream, F(77), wihT16, 2048, 1555, 1568);
  launch_cvt(stream, F(83), wih0_16, 1024, 2318, 2336);

  // ---- note_fc: 3 x (GEMM + bias + ReLU), f16 outputs chained ----
  launch_gemm(stream, Xp16, 0, 1, 96,  W0p, 96,  F(7),  nullptr, nullptr, Hfc1, 0, 1, 256, 0, 2048, 256, 96,  1);
  launch_gemm(stream, Hfc1, 0, 1, 256, W1p, 256, F(9),  nullptr, nullptr, Hfc2, 0, 1, 256, 0, 2048, 256, 256, 1);
  launch_gemm(stream, Hfc2, 0, 1, 256, W2p, 256, F(11), nullptr, nullptr, H16,  0, 1, 256, 0, 2048, 256, 256, 1);

  // ---- voice_net: 4 voices x 2 layers of bi-LSTM (H=128, T=512) ----
  // layer 0: input = H16 rows v::4 (strided A), output compact vbuf0 [(v)][t][256]
  for (int v = 0; v < 4; ++v)
    for (int d = 0; d < 2; ++d) {
      int base = d ? 16 : 12;  // l0 f / l0 b
      launch_gemm(stream, H16, v, 4, 256, vwih[d], 256, F(base + 2), F(base + 3),
                  Gih, nullptr, (long long)(v * 2 + d) * 512, 1, 512, 0, 512, 512, 256, 0);
    }
  lstm_scan_kernel<<<8, 512, 0, stream>>>(Gih, F(13), F(17), nullptr, vbuf0,
                                          512, 128, 256, 0, 0, 512, 1);
  // layer 1: input = vbuf0, output scattered into note_out cols [512,768) (row = v + 4t)
  for (int v = 0; v < 4; ++v)
    for (int d = 0; d < 2; ++d) {
      int base = d ? 24 : 20;
      launch_gemm(stream, vbuf0, (long long)v * 512, 1, 256, vwih[2 + d], 256, F(base + 2), F(base + 3),
                  Gih, nullptr, (long long)(v * 2 + d) * 512, 1, 512, 0, 512, 512, 256, 0);
    }
  lstm_scan_kernel<<<8, 512, 0, stream>>>(Gih, F(21), F(25), noteOutF, noteOut16,
                                          512, 128, 768, 512, 0, 1, 4);

  // ---- main note bi-LSTM (H=256, T=2048, 2 layers) ----
  for (int d = 0; d < 2; ++d) {
    int base = d ? 32 : 28;
    launch_gemm(stream, H16, 0, 1, 256, lwih0[d], 256, F(base + 2), F(base + 3),
                Gih, nullptr, (long long)d * 2048, 1, 1024, 0, 2048, 1024, 256, 0);
  }
  lstm_scan_kernel<<<2, 512, 0, stream>>>(Gih, F(29), F(33), nullptr, l0out16,
                                          2048, 256, 512, 0, 0, 0, 1);
  for (int d = 0; d < 2; ++d) {
    int base = d ? 40 : 36;
    launch_gemm(stream, l0out16, 0, 1, 512, lwih1[d], 512, F(base + 2), F(base + 3),
                Gih, nullptr, (long long)d * 2048, 1, 1024, 0, 2048, 1024, 512, 0);
  }
  lstm_scan_kernel<<<2, 512, 0, stream>>>(Gih, F(37), F(41), noteOutF, noteOut16,
                                          2048, 256, 768, 0, 0, 0, 1);

  // ---- beat attention (groups of 4 notes) ----
  launch_gemm(stream, noteOut16, 0, 1, 768, battnW16, 768, F(45), nullptr,
              attnA, nullptr, 0, 1, 768, 0, 2048, 768, 768, 2);
  group_attention_kernel<<<512, 256, 0, stream>>>(noteOutF, attnA, F(46), bnodesF, bnodes16, 768, 8, 4);

  // ---- beat bi-LSTM (H=512, T=512, 2 layers) ----
  for (int d = 0; d < 2; ++d) {
    int base = d ? 51 : 47;
    launch_gemm(stream, bnodes16, 0, 1, 768, bwih0[d], 768, F(base + 2), F(base + 3),
                Gih, nullptr, (long long)d * 512, 1, 2048, 0, 512, 2048, 768, 0);
  }
  lstm_scan_kernel<<<2, 512, 0, stream>>>(Gih, F(48), F(52), nullptr, beatl016,
                                          512, 512, 1024, 0, 0, 0, 1);
  for (int d = 0; d < 2; ++d) {
    int base = d ? 59 : 55;
    launch_gemm(stream, beatl016, 0, 1, 1024, bwih1[d], 1024, F(base + 2), F(base + 3),
                Gih, nullptr, (long long)d * 512, 1, 2048, 0, 512, 2048, 1024, 0);
  }
  lstm_scan_kernel<<<2, 512, 0, stream>>>(Gih, F(56), F(60), beatHF, beatH16,
                                          512, 512, 1024, 0, 0, 0, 1);

  // ---- measure attention + measure bi-LSTM (H=256, T=128, 1 layer) ----
  launch_gemm(stream, beatH16, 0, 1, 1024, mattnW16, 1024, F(64), nullptr,
              attnA, nullptr, 0, 1, 1024, 0, 512, 1024, 1024, 2);
  group_attention_kernel<<<128, 256, 0, stream>>>(beatHF, attnA, F(65), mnodesF, mnodes16, 1024, 8, 4);
  for (int d = 0; d < 2; ++d) {
    int base = d ? 70 : 66;
    launch_gemm(stream, mnodes16, 0, 1, 1024, mwih[d], 1024, F(base + 2), F(base + 3),
                Gih, nullptr, (long long)d * 128, 1, 1024, 0, 128, 1024, 1024, 0);
  }
  lstm_scan_kernel<<<2, 512, 0, stream>>>(Gih, F(67), F(71), measHF, nullptr,
                                          128, 256, 512, 0, 0, 0, 1);

  // ---- decoder static projections (WMMA), then persistent autoregressive scan ----
  {
    long long tn = 512LL * 1568;
    build_tempo_static_kernel<<<(int)((tn + 255) / 256), 256, 0, stream>>>(F(0), beatHF, measHF, tStat16);
    long long nn = 2048LL * 2336;
    build_note_static_kernel<<<(int)((nn + 255) / 256), 256, 0, stream>>>(F(0), noteOutF, beatHF, measHF, nStat16);
  }
  launch_gemm(stream, tStat16, 0, 1, 1568, wihT16, 1568, F(79), F(80),
              gstatT, nullptr, 0, 1, 2048, 0, 512, 2048, 1568, 0);
  launch_gemm(stream, nStat16, 0, 1, 2336, wih0_16, 2336, F(85), F(86),
              gstatN, nullptr, 0, 1, 1024, 0, 2048, 1024, 2336, 0);

  float* outTotal = (float*)d_out;                 // [1,2048,11]
  float* hiddenTotal = (float*)d_out + 2048LL * 11; // [1,2048,2304]
  decoder_scan_kernel<<<1, 1024, 0, stream>>>(
      gstatT, F(77), F(78), F(81), F(82),
      gstatN, F(83), F(84), F(87), F(88), F(89), F(90),
      F(91), F(92), F(74), F(75), F(76), outTotal);

  long long ht = 2048LL * 2304;
  hidden_total_kernel<<<(int)((ht + 255) / 256), 256, 0, stream>>>(
      noteOutF, beatHF, measHF, (const int*)d_in[2], (const int*)d_in[3], hiddenTotal);
}
